// Seq2SeqLSTM_89721866814192
// MI455X (gfx1250) — matrix operand order repaired and checked by reading, not compiled, not run on hardware
//
#include <hip/hip_runtime.h>

// ---------------------------------------------------------------------------
// Seq2Seq LSTM for MI455X (gfx1250, wave32, WMMA).
//
// v5: every GEMM operand is bf16. x is converted once to bf16 AND transposed
// to (T, B, F) so a timestep's A-fragment rows are 128B apart (cache-friendly);
// h state lives in bf16 (it is only ever a WMMA A-operand); weights are bf16,
// staged per-block into LDS via async DMA (global_load_async_to_lds_b128,
// ASYNCcnt + s_wait_asynccnt) and read with ds_load_b128. This removes all
// f32->bf16 cvts (and their WMMA->VALU WAR v_nop filler) from inner loops.
// c stays f32 for recurrence accuracy; f32 WMMA accumulation.
// Host loop: ~1200 stream-ordered launches; stream order = recurrence barrier.
// ---------------------------------------------------------------------------

#define USE_ASYNC_LDS_ASM 1

static constexpr int Bsz   = 256;
static constexpr int TINs  = 512;
static constexpr int Fin   = 64;
static constexpr int Hd    = 512;
static constexpr int TOUTs = 64;

typedef __attribute__((ext_vector_type(16))) __bf16          v16bf;
typedef __attribute__((ext_vector_type(8)))  float           v8f;
typedef __attribute__((ext_vector_type(8)))  unsigned short  ushortx8;

#ifndef __has_builtin
#define __has_builtin(x) 0
#endif

__device__ __forceinline__ float sigf(float x) { return 1.0f / (1.0f + __expf(-x)); }

// Copy 16B global -> LDS via GLOBAL_LOAD_ASYNC_TO_LDS_B128 (ISA 15.18.3 op 98):
// VDST = LDS byte offset, VADDR = 64-bit global address (GV mode); ASYNCcnt.
__device__ __forceinline__ void stage16(unsigned short* dst, const unsigned short* src) {
#if USE_ASYNC_LDS_ASM
  unsigned ldsOff = (unsigned)(unsigned long long)dst;   // LDS aperture: addr[31:0]
  asm volatile("global_load_async_to_lds_b128 %0, %1, off"
               :
               : "v"(ldsOff), "v"(src)
               : "memory");
#else
  *(ushortx8*)dst = *(const ushortx8*)src;
#endif
}

__device__ __forceinline__ void async_join() {
#if USE_ASYNC_LDS_ASM
#if __has_builtin(__builtin_amdgcn_s_wait_asynccnt)
  __builtin_amdgcn_s_wait_asynccnt(0);
#else
  asm volatile("s_wait_asynccnt 0x0" ::: "memory");
#endif
#endif
}

// A fragment 16x32 from bf16 rows: two raw 16B loads per lane, zero VALU.
// ISA 16-bit A layout: lane -> M = lane&15; lanes 0-15 K halves {0..7,16..23},
// lanes 16-31 K halves {8..15,24..31}.
__device__ __forceinline__ v16bf load_a_bf16(const unsigned short* __restrict__ base,
                                             long srow, int b0, int k0, int lane) {
  const int m  = lane & 15;
  const int kb = (lane & 16) ? 8 : 0;
  const unsigned short* p = base + (long)(b0 + m) * srow + k0 + kb;
  struct P { ushortx8 lo, hi; } pk;
  pk.lo = *(const ushortx8*)(p);        // K = k0+kb    .. +7
  pk.hi = *(const ushortx8*)(p + 16);   // K = k0+16+kb .. +7
  return __builtin_bit_cast(v16bf, pk);
}

// B fragment (32x16 = W.T tile) from LDS: two 8-bf16 halves per lane
// (2x ds_load_b128), K = k0+kb..+7 and k0+16+kb..+7. Row r = g*16+n, stride K.
__device__ __forceinline__ v16bf load_b_lds(const unsigned short* __restrict__ Wl,
                                            int K, int g, int k0, int lane) {
  const int n  = lane & 15;
  const int kb = (lane & 16) ? 8 : 0;
  const unsigned short* p = Wl + (long)(g * 16 + n) * K + k0 + kb;
  struct P { ushortx8 lo, hi; } pk;
  pk.lo = *(const ushortx8*)(p);
  pk.hi = *(const ushortx8*)(p + 16);
  return __builtin_bit_cast(v16bf, pk);
}

// GEMM over LDS weights. K % 64 == 0. Ping-pong A fragments: the loop-carried
// value is always produced by a VMEM load (no register copy, no VALU hazard).
__device__ __forceinline__ void gemm_part(v8f acc[4],
                                          const unsigned short* __restrict__ Abase,
                                          long srow, int b0,
                                          const unsigned short* __restrict__ Wl,
                                          int K, int lane) {
  v16bf a0 = load_a_bf16(Abase, srow, b0, 0, lane);
  for (int k0 = 0; k0 < K; k0 += 64) {
    v16bf a1 = load_a_bf16(Abase, srow, b0, k0 + 32, lane);
#pragma unroll
    for (int g = 0; g < 4; ++g) {
      v16bf bm = load_b_lds(Wl, K, g, k0, lane);
      acc[g] = __builtin_amdgcn_wmma_f32_16x16x32_bf16(
          false, a0, false, bm, (short)0, acc[g], false, false);
    }
    a0 = load_a_bf16(Abase, srow, b0, (k0 + 64 < K) ? (k0 + 64) : 0, lane);
#pragma unroll
    for (int g = 0; g < 4; ++g) {
      v16bf bm = load_b_lds(Wl, K, g, k0 + 32, lane);
      acc[g] = __builtin_amdgcn_wmma_f32_16x16x32_bf16(
          false, a1, false, bm, (short)0, acc[g], false, false);
    }
  }
}

// One LSTM layer-step. Block = 8 waves sharing one jtile; wave -> 16x16
// (batch x hidden) patch of all four gates. Grid 64 blocks x 256 threads.
// xin: bf16 (B,Kx)-strided activations when Kx>=32; f32 (B,) scalar when Kx==1.
__global__ __launch_bounds__(256) void lstm_step_kernel(
    const void* __restrict__ xin, int Kx, long srowx,
    const unsigned short* __restrict__ WihB,   // bf16 (4H, Kx), when Kx >= 32
    const float* __restrict__ Wih1,            // f32  (4H, 1),  when Kx == 1
    const unsigned short* __restrict__ WhhB,   // bf16 (4H, H)
    const float* __restrict__ bih, const float* __restrict__ bhh,
    const unsigned short* __restrict__ hin,    // bf16 (B, H)
    unsigned short* __restrict__ hout,         // bf16 (B, H)  ping-pong
    float* __restrict__ cst)                   // f32  (B, H)  in place
{
  extern __shared__ unsigned short ldsW[];

  const int tid   = threadIdx.x;
  const int lane  = tid & 31;
  const int wid   = tid >> 5;
  const int jtile = blockIdx.x >> 1;
  const int btile = ((blockIdx.x & 1) << 3) + wid;
  const int b0 = btile * 16;
  const int j0 = jtile * 16;

  // ---- async staging of this block's weight tiles into LDS ----------------
  const int wOff = (Kx >= 32) ? 64 * Kx : 0;        // Whh tile offset (elems)
  if (Kx >= 32) {
    const int gpr = Kx >> 3;                        // 16B granules per row
    for (int i = tid; i < 64 * gpr; i += 256) {
      const int r = i / gpr, cg = i - r * gpr;
      const int g = r >> 4, n = r & 15;
      stage16(ldsW + (long)r * Kx + cg * 8,
              WihB + (long)(g * Hd + j0 + n) * Kx + cg * 8);
    }
  }
  {
    const int gpr = Hd >> 3;
    for (int i = tid; i < 64 * gpr; i += 256) {
      const int r = i / gpr, cg = i - r * gpr;
      const int g = r >> 4, n = r & 15;
      stage16(ldsW + wOff + (long)r * Hd + cg * 8,
              WhhB + (long)(g * Hd + j0 + n) * Hd + cg * 8);
    }
  }

  // ---- prefetch epilogue operands while staging is in flight --------------
  const int n   = lane & 15;
  const int hi8 = (lane & 16) ? 8 : 0;
  const int jn  = j0 + n;
  float bsum[4];
#pragma unroll
  for (int g = 0; g < 4; ++g) bsum[g] = bih[g * Hd + jn] + bhh[g * Hd + jn];
  float cold[8];
#pragma unroll
  for (int r = 0; r < 8; ++r) cold[r] = cst[(long)(b0 + r + hi8) * Hd + jn];

  async_join();
  __syncthreads();

  v8f acc[4] = {};   // gate order: i, f, g, o

  if (Kx >= 32)
    gemm_part(acc, (const unsigned short*)xin, srowx, b0, ldsW, Kx, lane); // x@Wih.T
  gemm_part(acc, hin, (long)Hd, b0, ldsW + wOff, Hd, lane);                // h@Whh.T

  // ---- rank-1 input update for dec0 (input size 1, f32 prev) --------------
  if (Kx == 1) {
    const float* xs = (const float*)xin;
    float w[4];
#pragma unroll
    for (int g = 0; g < 4; ++g) w[g] = Wih1[g * Hd + jn];
#pragma unroll
    for (int r = 0; r < 8; ++r) {
      float xv = xs[b0 + r + hi8];
#pragma unroll
      for (int g = 0; g < 4; ++g) acc[g][r] += xv * w[g];
    }
  }

  // ---- bias + fused LSTM cell update --------------------------------------
  // C/D layout: VGPR r holds M = r (lanes 0-15) / r+8 (lanes 16-31), N = lane&15.
#pragma unroll
  for (int r = 0; r < 8; ++r) {
    long idx = (long)(b0 + r + hi8) * Hd + jn;
    float iv = sigf(acc[0][r] + bsum[0]);
    float fv = sigf(acc[1][r] + bsum[1]);
    float gv = tanhf(acc[2][r] + bsum[2]);
    float ov = sigf(acc[3][r] + bsum[3]);
    float cn = fv * cold[r] + iv * gv;
    cst[idx] = cn;
    __bf16 hb = (__bf16)(ov * tanhf(cn));
    hout[idx] = __builtin_bit_cast(unsigned short, hb);
  }
}

// fp32 -> bf16 weight conversion (once per call; amortized over 512 steps)
__global__ void cvt_f32_bf16(const float* __restrict__ s, unsigned short* __restrict__ d, int n) {
  int i = blockIdx.x * blockDim.x + threadIdx.x;
  if (i < n) {
    __bf16 v = (__bf16)s[i];
    d[i] = __builtin_bit_cast(unsigned short, v);
  }
}

// x (B, T, F) f32 -> x_bf (T, B, F) bf16: one-time convert + transpose so a
// timestep's 16 A-fragment rows are contiguous (128B apart) instead of 64KB.
__global__ void cvt_x_kernel(const float* __restrict__ src, unsigned short* __restrict__ dst) {
  long i = (long)blockIdx.x * blockDim.x + threadIdx.x;
  if (i < (long)Bsz * TINs * Fin) {
    int b   = (int)(i / (TINs * Fin));
    int rem = (int)(i - (long)b * (TINs * Fin));
    int t = rem / Fin, k = rem - t * Fin;
    __bf16 v = (__bf16)src[i];
    dst[((long)t * Bsz + b) * Fin + k] = __builtin_bit_cast(unsigned short, v);
  }
}

// Zero h (bf16) and c (f32) state, init decoder feedback prev = x[:, -1, -1].
__global__ void init_state(unsigned short* h0a, unsigned short* h0b,
                           unsigned short* h1a, unsigned short* h1b,
                           float* c0, float* c1, float* prev,
                           const float* __restrict__ x) {
  int i = blockIdx.x * blockDim.x + threadIdx.x;
  if (i < Bsz * Hd) {
    h0a[i] = 0; h0b[i] = 0; h1a[i] = 0; h1b[i] = 0;
    c0[i] = 0.f; c1[i] = 0.f;
  }
  if (i < Bsz) prev[i] = x[(long)i * TINs * Fin + (long)(TINs - 1) * Fin + (Fin - 1)];
}

// step = h1 @ proj_W.T + proj_b ; h1 is bf16. Writes output column t + prev.
__global__ __launch_bounds__(256) void proj_kernel(
    const unsigned short* __restrict__ h1, const float* __restrict__ pw,
    const float* __restrict__ pb, float* __restrict__ prev,
    float* __restrict__ out, int t) {
  int b = threadIdx.x;
  const unsigned short* row = h1 + (long)b * Hd;
  float s = 0.f;
  for (int j = 0; j < Hd; ++j) {
    __bf16 hv = __builtin_bit_cast(__bf16, row[j]);
    s += (float)hv * pw[j];
  }
  s += pb[0];
  prev[b] = s;
  out[(long)b * TOUTs + t] = s;
}

extern "C" void kernel_launch(void* const* d_in, const int* in_sizes, int n_in,
                              void* d_out, int out_size, void* d_ws, size_t ws_size,
                              hipStream_t stream) {
  const float* x        = (const float*)d_in[0];
  const float* enc0_Wih = (const float*)d_in[2];
  const float* enc0_Whh = (const float*)d_in[3];
  const float* enc0_bih = (const float*)d_in[4];
  const float* enc0_bhh = (const float*)d_in[5];
  const float* enc1_Wih = (const float*)d_in[6];
  const float* enc1_Whh = (const float*)d_in[7];
  const float* enc1_bih = (const float*)d_in[8];
  const float* enc1_bhh = (const float*)d_in[9];
  const float* dec0_Wih = (const float*)d_in[10];
  const float* dec0_Whh = (const float*)d_in[11];
  const float* dec0_bih = (const float*)d_in[12];
  const float* dec0_bhh = (const float*)d_in[13];
  const float* dec1_Wih = (const float*)d_in[14];
  const float* dec1_Whh = (const float*)d_in[15];
  const float* dec1_bih = (const float*)d_in[16];
  const float* dec1_bhh = (const float*)d_in[17];
  const float* projW    = (const float*)d_in[18];
  const float* projb    = (const float*)d_in[19];
  float* out = (float*)d_out;

  // ---- workspace carving (256B aligned), ~31 MB ----
  char* ws = (char*)d_ws;
  size_t off = 0;
  auto carve = [&](size_t bytes) -> char* {
    char* p = ws + off;
    off = (off + bytes + 255) & ~(size_t)255;
    return p;
  };
  unsigned short* w_e0_ih = (unsigned short*)carve((size_t)4 * Hd * Fin * 2);
  unsigned short* w_e0_hh = (unsigned short*)carve((size_t)4 * Hd * Hd * 2);
  unsigned short* w_e1_ih = (unsigned short*)carve((size_t)4 * Hd * Hd * 2);
  unsigned short* w_e1_hh = (unsigned short*)carve((size_t)4 * Hd * Hd * 2);
  unsigned short* w_d0_hh = (unsigned short*)carve((size_t)4 * Hd * Hd * 2);
  unsigned short* w_d1_ih = (unsigned short*)carve((size_t)4 * Hd * Hd * 2);
  unsigned short* w_d1_hh = (unsigned short*)carve((size_t)4 * Hd * Hd * 2);
  unsigned short* x_bf    = (unsigned short*)carve((size_t)Bsz * TINs * Fin * 2);
  unsigned short* h0[2]; unsigned short* h1[2];
  h0[0] = (unsigned short*)carve((size_t)Bsz * Hd * 2);
  h0[1] = (unsigned short*)carve((size_t)Bsz * Hd * 2);
  h1[0] = (unsigned short*)carve((size_t)Bsz * Hd * 2);
  h1[1] = (unsigned short*)carve((size_t)Bsz * Hd * 2);
  float* c0   = (float*)carve((size_t)Bsz * Hd * 4);
  float* c1   = (float*)carve((size_t)Bsz * Hd * 4);
  float* prev = (float*)carve((size_t)Bsz * 4);

  // ---- one-time conversions (bf16 weights + transposed bf16 x) ----
  auto cvt = [&](const float* s, unsigned short* d, int n) {
    cvt_f32_bf16<<<(n + 255) / 256, 256, 0, stream>>>(s, d, n);
  };
  cvt(enc0_Wih, w_e0_ih, 4 * Hd * Fin);
  cvt(enc0_Whh, w_e0_hh, 4 * Hd * Hd);
  cvt(enc1_Wih, w_e1_ih, 4 * Hd * Hd);
  cvt(enc1_Whh, w_e1_hh, 4 * Hd * Hd);
  cvt(dec0_Whh, w_d0_hh, 4 * Hd * Hd);
  cvt(dec1_Wih, w_d1_ih, 4 * Hd * Hd);
  cvt(dec1_Whh, w_d1_hh, 4 * Hd * Hd);
  {
    long nx = (long)Bsz * TINs * Fin;
    cvt_x_kernel<<<(unsigned)((nx + 255) / 256), 256, 0, stream>>>(x, x_bf);
  }

  init_state<<<(Bsz * Hd + 255) / 256, 256, 0, stream>>>(
      h0[0], h0[1], h1[0], h1[1], c0, c1, prev, x);

  // Dynamic LDS: Wih tile (64 x Kx) + Whh tile (64 x Hd), bf16.
  const size_t shm_enc0 = (size_t)(64 * Fin + 64 * Hd) * 2;   //  72 KB
  const size_t shm_full = (size_t)(64 * Hd + 64 * Hd) * 2;    // 128 KB
  const size_t shm_dec0 = (size_t)(64 * Hd) * 2;              //  64 KB

  // ---- encoder: 512 steps x 2 layers, stream order = recurrence barrier ----
  for (int t = 0; t < TINs; ++t) {
    int cur = t & 1, nxt = cur ^ 1;
    lstm_step_kernel<<<64, 256, shm_enc0, stream>>>(
        x_bf + (long)t * Bsz * Fin, Fin, (long)Fin,
        w_e0_ih, nullptr, w_e0_hh, enc0_bih, enc0_bhh, h0[cur], h0[nxt], c0);
    lstm_step_kernel<<<64, 256, shm_full, stream>>>(
        h0[nxt], Hd, (long)Hd,
        w_e1_ih, nullptr, w_e1_hh, enc1_bih, enc1_bhh, h1[cur], h1[nxt], c1);
  }

  // ---- decoder: 64 steps x 2 layers + projection feedback ----
  // TIN is even, so final encoder state lands in index 0 = (t=0)&1.
  for (int t = 0; t < TOUTs; ++t) {
    int cur = t & 1, nxt = cur ^ 1;
    lstm_step_kernel<<<64, 256, shm_dec0, stream>>>(
        prev, 1, 1,
        nullptr, dec0_Wih, w_d0_hh, dec0_bih, dec0_bhh, h0[cur], h0[nxt], c0);
    lstm_step_kernel<<<64, 256, shm_full, stream>>>(
        h0[nxt], Hd, (long)Hd,
        w_d1_ih, nullptr, w_d1_hh, dec1_bih, dec1_bhh, h1[cur], h1[nxt], c1);
    proj_kernel<<<1, 256, 0, stream>>>(h1[nxt], projW, projb, prev, out, t);
  }
}
